// RoPEMultiHeadSelfAttention_89773406421446
// MI455X (gfx1250) — matrix operand order repaired and checked by reading, not compiled, not run on hardware
//
#include <hip/hip_runtime.h>
#include <hip/hip_bf16.h>
#include <math.h>

// ---------------------------------------------------------------------------
// RoPE multi-head self-attention for MI455X (gfx1250, wave32, WMMA + TDM).
// b=4, s=2048, d_model=1024, heads=16, head_dim=64.
// Matmuls on v_wmma_f32_16x16x32_f16 (f32 accumulate); flash-attention tiling
// keeps the 2048x2048 logits in VGPRs/LDS; K/A tiles staged to LDS by the
// Tensor Data Mover (tensor_load_to_lds + s_wait_tensorcnt) when available.
// ---------------------------------------------------------------------------

typedef __attribute__((ext_vector_type(16))) _Float16 v16h;
typedef __attribute__((ext_vector_type(8)))  float    v8f;
typedef unsigned int u32x4 __attribute__((ext_vector_type(4)));
typedef int          i32x4 __attribute__((ext_vector_type(4)));
typedef int          i32x8 __attribute__((ext_vector_type(8)));

#define D_MODEL   1024
#define NUM_HEADS 16
#define HEAD_DIM  64
#define SEQ       2048
#define BATCH     4
#define ROWS      (BATCH * SEQ)          // 8192
#define QKV_COLS  (3 * D_MODEL)          // 3072

// ---- Tensor Data Mover availability ---------------------------------------
#if defined(__HIP_DEVICE_COMPILE__) && defined(__gfx1250__) && \
    __has_builtin(__builtin_amdgcn_tensor_load_to_lds)
#define USE_TDM 1
#else
#define USE_TDM 0
#endif

#if USE_TDM
// 2D f16 tile load, global -> LDS, per CDNA5 ISA.
// tile_d0 = contiguous elements per row, tile_d1 = rows,
// stride_elems = elements between consecutive rows in global memory.
// Tensor dims are set equal to the tile (always a full interior tile here).
__device__ __forceinline__ void tdm_load_tile_f16(unsigned lds_off,
                                                  const _Float16* g,
                                                  unsigned tile_d0,
                                                  unsigned tile_d1,
                                                  unsigned long long stride_elems) {
    unsigned long long ga = (unsigned long long)(size_t)g;
    u32x4 g0;
    g0.x = 1u;                                           // count=1, user D#
    g0.y = lds_off;                                      // lds_addr (bytes)
    g0.z = (unsigned)ga;                                 // global_addr[31:0]
    g0.w = (unsigned)((ga >> 32) & 0x01ffffffu) | (2u << 30); // [56:32] | type=2
    i32x8 g1;
    g1[0] = 0x00010000;                                  // data_size=1 (2 bytes)
    g1[1] = (int)((tile_d0 & 0xffffu) << 16);            // tensor_dim0[15:0]
    g1[2] = (int)((tile_d1 & 0xffffu) << 16);            // tensor_dim1[15:0]
    g1[3] = (int)((tile_d0 & 0xffffu) << 16);            // tile_dim0
    g1[4] = (int)(tile_d1 & 0xffffu);                    // tile_dim1 (tile_dim2=0)
    g1[5] = (int)(unsigned)(stride_elems & 0xffffffffu); // dim0_stride[31:0]
    g1[6] = (int)(unsigned)((stride_elems >> 32) & 0xffffu); // dim0_stride[47:32]
    g1[7] = 0;                                           // dim1_stride (unused, 2D)
    i32x4 z4 = {0, 0, 0, 0};
#if __clang_major__ >= 23
    i32x8 z8 = {0, 0, 0, 0, 0, 0, 0, 0};
    __builtin_amdgcn_tensor_load_to_lds(g0, g1, z4, z4, z8, 0);
#else
    __builtin_amdgcn_tensor_load_to_lds(g0, g1, z4, z4, 0);
#endif
}

__device__ __forceinline__ void tdm_wait() {
#if __has_builtin(__builtin_amdgcn_s_wait_tensorcnt)
    __builtin_amdgcn_s_wait_tensorcnt(0);
#else
    asm volatile("s_wait_tensorcnt 0x0" ::: "memory");
#endif
}
#endif  // USE_TDM

// ---- fragment assembly helpers (per ISA 7.12.2 layouts) -------------------
// A 16x32 f16: lane<16 holds row=lane, K = {0..7} then {16..23};
//              lane>=16 holds row=lane-16, K = {8..15} then {24..31}.
__device__ inline v16h make_frag_a(const _Float16* p0, const _Float16* p1) {
    v16h r;
#pragma unroll
    for (int i = 0; i < 8; ++i) r[i] = p0[i];
#pragma unroll
    for (int i = 0; i < 8; ++i) r[8 + i] = p1[i];
    return r;
}
// B 32x16 f16: lane<16 holds col=lane, K = {0..7} then {16..23};
//              lane>=16 holds col=lane-16, K = {8..15} then {24..31}.
__device__ inline v16h make_frag_b(const _Float16* p) {
    v16h r;
#pragma unroll
    for (int i = 0; i < 8; ++i) r[i] = p[i];
#pragma unroll
    for (int i = 0; i < 8; ++i) r[8 + i] = p[16 + i];
    return r;
}

// ---------------------------------------------------------------------------
// f32 -> f16 conversion (grid-stride, memory bound, trivial share of runtime)
// ---------------------------------------------------------------------------
__global__ void cvt_f32_to_f16(const float* __restrict__ in,
                               _Float16* __restrict__ out, size_t n) {
    size_t i = (size_t)blockIdx.x * blockDim.x + threadIdx.x;
    size_t stride = (size_t)gridDim.x * blockDim.x;
    for (; i < n; i += stride) out[i] = (_Float16)in[i];
}

// ---------------------------------------------------------------------------
// Tiled WMMA GEMM:  C[M,N] = A[M,K] @ B[K,N],  f16 in, OutT out, f32 accum.
// Block = 256 threads (8 waves). Block tile 64x128, wave tile 32x32
// (2x2 of 16x16 WMMA). K loop in steps of 32 through LDS.
// A tile staged by TDM; B tile staged manually (needs transpose).
// ---------------------------------------------------------------------------
template <typename OutT>
__global__ __launch_bounds__(256)
void gemm_f16_wmma(const _Float16* __restrict__ A,
                   const _Float16* __restrict__ B,
                   OutT* __restrict__ C, int M, int N, int K) {
    __shared__ _Float16 As[64][32];   // row-major A tile        (4 KB)
    __shared__ _Float16 Bs[128][32];  // B tile transposed [n][k] (8 KB)

    const int tid    = threadIdx.x;
    const int wave   = tid >> 5;          // 0..7
    const int lane   = tid & 31;
    const int laneHi = lane >> 4;         // 0/1
    const int lane15 = lane & 15;
    const int m0 = blockIdx.x * 64;
    const int n0 = blockIdx.y * 128;
    const int wm = (wave & 1) * 32;       // wave M offset in block tile
    const int wn = (wave >> 1) * 32;      // wave N offset in block tile

    v8f acc[2][2] = {};

    for (int kk = 0; kk < K; kk += 32) {
        __syncthreads();
#if USE_TDM
        if (wave == 0)   // DMA the 64x32 A tile while threads stage B
            tdm_load_tile_f16((unsigned)(size_t)&As[0][0],
                              A + (size_t)m0 * K + kk, 32u, 64u,
                              (unsigned long long)K);
#else
        {   // A tile 64x32: 8 f16 per thread, contiguous 16B global loads
            int r = tid >> 2, c = (tid & 3) * 8;
            const _Float16* g = A + (size_t)(m0 + r) * K + kk + c;
#pragma unroll
            for (int i = 0; i < 8; ++i) As[r][c + i] = g[i];
        }
#endif
        {   // B tile 32x128: 16 f16 per thread, stored transposed for
            // contiguous per-lane B-fragment reads
            int r = tid >> 3, c0 = (tid & 7) * 16;
            const _Float16* g = B + (size_t)(kk + r) * N + n0 + c0;
#pragma unroll
            for (int i = 0; i < 16; ++i) Bs[c0 + i][r] = g[i];
        }
#if USE_TDM
        if (wave == 0) tdm_wait();
#endif
        __syncthreads();

#pragma unroll
        for (int mi = 0; mi < 2; ++mi) {
            const int arow = wm + mi * 16 + lane15;
            const int ak   = laneHi * 8;
            v16h a = make_frag_a(&As[arow][ak], &As[arow][16 + ak]);
#pragma unroll
            for (int ni = 0; ni < 2; ++ni) {
                const int bcol = wn + ni * 16 + lane15;
                v16h b = make_frag_b(&Bs[bcol][laneHi * 8]);
                acc[mi][ni] = __builtin_amdgcn_wmma_f32_16x16x32_f16(
                    false, a, false, b, (short)0, acc[mi][ni], false, false);
            }
        }
    }

    // C/D layout: row = vgpr + 8*laneHi, col = lane15
#pragma unroll
    for (int mi = 0; mi < 2; ++mi)
#pragma unroll
        for (int ni = 0; ni < 2; ++ni) {
            const int col = n0 + wn + ni * 16 + lane15;
#pragma unroll
            for (int j = 0; j < 8; ++j) {
                const int row = m0 + wm + mi * 16 + j + laneHi * 8;
                C[(size_t)row * N + col] = (OutT)acc[mi][ni][j];
            }
        }
}

// ---------------------------------------------------------------------------
// RoPE applied in place to the q and k sections of qkv (f16).
// One thread per (row, head, half-dim index).
// ---------------------------------------------------------------------------
__global__ void rope_kernel(_Float16* __restrict__ qkv) {
    const int total = ROWS * NUM_HEADS * (HEAD_DIM / 2);
    int idx = blockIdx.x * blockDim.x + threadIdx.x;
    if (idx >= total) return;
    const int i   = idx & 31;           // 0..31  (half of head_dim)
    const int h   = (idx >> 5) & 15;
    const int row = idx >> 9;
    const int pos = row & (SEQ - 1);    // row % 2048 (rows are b-major)

    // inv_freq = 10000^(-i/32) = exp(-i * ln(10000)/32)
    const float inv = __expf(-(float)i * 0.28782313662425574f);
    const float th  = (float)pos * inv;
    const float c = __cosf(th), s = __sinf(th);

    _Float16* q = qkv + (size_t)row * QKV_COLS + h * HEAD_DIM;
    _Float16* k = q + D_MODEL;

    float q1 = (float)q[i], q2 = (float)q[i + 32];
    q[i]      = (_Float16)(q1 * c - q2 * s);
    q[i + 32] = (_Float16)(q2 * c + q1 * s);
    float k1 = (float)k[i], k2 = (float)k[i + 32];
    k[i]      = (_Float16)(k1 * c - k2 * s);
    k[i + 32] = (_Float16)(k2 * c + k1 * s);
}

// ---------------------------------------------------------------------------
// Flash attention: grid = (b*h = 64, seq/64 = 32), block = 128 (4 waves).
// Each wave owns 16 query rows; block processes 64 q-rows, streaming keys in
// 64-wide chunks with online softmax. K tile staged by TDM; V tile staged
// manually transposed. Logits stay in VGPRs; P goes through LDS only to
// re-shape C-layout -> A-layout for the P@V WMMA.
// ---------------------------------------------------------------------------
__global__ __launch_bounds__(128)
void flash_attn(const _Float16* __restrict__ qkv,
                const float* __restrict__ mask,
                _Float16* __restrict__ ctx) {
    const int bh    = blockIdx.x;
    const int b     = bh >> 4;
    const int h     = bh & 15;
    const int qtile = blockIdx.y * 64;
    const int tid   = threadIdx.x;
    const int wave  = tid >> 5;          // 0..3
    const int lane  = tid & 31;
    const int laneHi = lane >> 4;
    const int lane15 = lane & 15;

    __shared__ _Float16 Ks[64][64];      // [key][dim]            8 KB
    __shared__ _Float16 Vs[64][64];      // transposed [dim][key] 8 KB
    __shared__ _Float16 Ps[4][16][64];   // per-wave P tile       8 KB

    const size_t rs = QKV_COLS;
    const _Float16* qbase = qkv + (size_t)(b * SEQ + qtile + wave * 16) * rs + h * HEAD_DIM;
    const _Float16* kbase = qkv + (size_t)(b * SEQ) * rs + D_MODEL + h * HEAD_DIM;
    const _Float16* vbase = kbase + D_MODEL;
    const float*    mrow  = mask + b * SEQ;

    // Load the wave's 16 q rows into two A fragments (K = 0..31, 32..63),
    // folding the 1/sqrt(64) = 0.125 logit scale into q.
    v16h a0, a1;
    {
        const _Float16* qp = qbase + (size_t)lane15 * rs;
        const int o = laneHi * 8;
#pragma unroll
        for (int i = 0; i < 8; ++i) a0[i]     = (_Float16)((float)qp[o + i]      * 0.125f);
#pragma unroll
        for (int i = 0; i < 8; ++i) a0[8 + i] = (_Float16)((float)qp[16 + o + i] * 0.125f);
#pragma unroll
        for (int i = 0; i < 8; ++i) a1[i]     = (_Float16)((float)qp[32 + o + i] * 0.125f);
#pragma unroll
        for (int i = 0; i < 8; ++i) a1[8 + i] = (_Float16)((float)qp[48 + o + i] * 0.125f);
    }

    float m[8], l[8];
#pragma unroll
    for (int j = 0; j < 8; ++j) { m[j] = -1e30f; l[j] = 0.0f; }
    v8f o_acc[4] = {};

    for (int kc = 0; kc < SEQ; kc += 64) {
        __syncthreads();   // previous chunk's K/V reads finished
#if USE_TDM
        if (wave == 0)     // DMA the 64x64 K tile (row-major)
            tdm_load_tile_f16((unsigned)(size_t)&Ks[0][0],
                              kbase + (size_t)kc * rs, 64u, 64u,
                              (unsigned long long)rs);
        {   // V tile transposed manually (TDM cannot transpose): 1 row / 2 thr
            const int r  = tid >> 1;
            const int c0 = (tid & 1) * 32;
            const _Float16* vg = vbase + (size_t)(kc + r) * rs + c0;
#pragma unroll
            for (int i = 0; i < 32; ++i) Vs[c0 + i][r] = vg[i];
        }
        if (wave == 0) tdm_wait();
#else
        {   // cooperative K/V tile load: 2 threads per key row
            const int r  = tid >> 1;
            const int c0 = (tid & 1) * 32;
            const _Float16* kg = kbase + (size_t)(kc + r) * rs + c0;
            const _Float16* vg = vbase + (size_t)(kc + r) * rs + c0;
#pragma unroll
            for (int i = 0; i < 32; ++i) Ks[r][c0 + i] = kg[i];
#pragma unroll
            for (int i = 0; i < 32; ++i) Vs[c0 + i][r] = vg[i];
        }
#endif
        __syncthreads();

        // ---- S = q @ k^T  (four 16x16 tiles, two K-steps each) ----
        v8f s[4];
#pragma unroll
        for (int nt = 0; nt < 4; ++nt) {
            const int key = nt * 16 + lane15;
            v16h b0 = make_frag_b(&Ks[key][laneHi * 8]);
            v16h b1 = make_frag_b(&Ks[key][32 + laneHi * 8]);
            v8f c = {};
            c = __builtin_amdgcn_wmma_f32_16x16x32_f16(false, a0, false, b0,
                                                       (short)0, c, false, false);
            c = __builtin_amdgcn_wmma_f32_16x16x32_f16(false, a1, false, b1,
                                                       (short)0, c, false, false);
            const float mv = (1.0f - mrow[kc + nt * 16 + lane15]) * -1e9f;
#pragma unroll
            for (int j = 0; j < 8; ++j) s[nt][j] = c[j] + mv;
        }

        // ---- online softmax: row max via xor-shuffle within 16-lane half ----
        float mn[8];
#pragma unroll
        for (int j = 0; j < 8; ++j) {
            float v = fmaxf(fmaxf(s[0][j], s[1][j]), fmaxf(s[2][j], s[3][j]));
#pragma unroll
            for (int x = 1; x < 16; x <<= 1) v = fmaxf(v, __shfl_xor(v, x, 32));
            mn[j] = fmaxf(m[j], v);
        }
        float rowsum[8];
#pragma unroll
        for (int j = 0; j < 8; ++j) {
            const float sf = __expf(m[j] - mn[j]);
            m[j] = mn[j];
            l[j] *= sf;
#pragma unroll
            for (int nt = 0; nt < 4; ++nt) o_acc[nt][j] *= sf;
            float acc = 0.0f;
#pragma unroll
            for (int nt = 0; nt < 4; ++nt) {
                const float p = __expf(s[nt][j] - mn[j]);
                s[nt][j] = p;
                acc += p;
            }
            rowsum[j] = acc;
        }
#pragma unroll
        for (int j = 0; j < 8; ++j) {
            float v = rowsum[j];
#pragma unroll
            for (int x = 1; x < 16; x <<= 1) v += __shfl_xor(v, x, 32);
            l[j] += v;
        }

        // ---- stage P (C-layout -> row-major f16 in LDS) ----
#pragma unroll
        for (int nt = 0; nt < 4; ++nt)
#pragma unroll
            for (int j = 0; j < 8; ++j)
                Ps[wave][j + 8 * laneHi][nt * 16 + lane15] = (_Float16)s[nt][j];
        __syncthreads();

        // ---- O += P @ V ----
        v16h pa0 = make_frag_a(&Ps[wave][lane15][laneHi * 8],
                               &Ps[wave][lane15][16 + laneHi * 8]);
        v16h pa1 = make_frag_a(&Ps[wave][lane15][32 + laneHi * 8],
                               &Ps[wave][lane15][48 + laneHi * 8]);
#pragma unroll
        for (int nt = 0; nt < 4; ++nt) {
            const int dim = nt * 16 + lane15;
            v16h b0 = make_frag_b(&Vs[dim][laneHi * 8]);
            v16h b1 = make_frag_b(&Vs[dim][32 + laneHi * 8]);
            o_acc[nt] = __builtin_amdgcn_wmma_f32_16x16x32_f16(
                false, pa0, false, b0, (short)0, o_acc[nt], false, false);
            o_acc[nt] = __builtin_amdgcn_wmma_f32_16x16x32_f16(
                false, pa1, false, b1, (short)0, o_acc[nt], false, false);
        }
    }

    // ---- normalize and write context in [b,s,h*64+dim] layout (f16) ----
    _Float16* cb = ctx + (size_t)(b * SEQ + qtile + wave * 16) * D_MODEL + h * HEAD_DIM;
#pragma unroll
    for (int nt = 0; nt < 4; ++nt) {
        const int dim = nt * 16 + lane15;
#pragma unroll
        for (int j = 0; j < 8; ++j) {
            const int r = j + 8 * laneHi;
            const float inv = l[j] > 0.0f ? 1.0f / l[j] : 0.0f;
            cb[(size_t)r * D_MODEL + dim] = (_Float16)(o_acc[nt][j] * inv);
        }
    }
}

// ---------------------------------------------------------------------------
// Launch: convert -> QKV GEMM -> RoPE -> flash attention -> out GEMM
// ---------------------------------------------------------------------------
extern "C" void kernel_launch(void* const* d_in, const int* in_sizes, int n_in,
                              void* d_out, int out_size, void* d_ws, size_t ws_size,
                              hipStream_t stream) {
    const float* x    = (const float*)d_in[0];   // [4,2048,1024]
    const float* mask = (const float*)d_in[1];   // [4,2048]
    const float* Wqkv = (const float*)d_in[2];   // [1024,3072]
    const float* Wout = (const float*)d_in[3];   // [1024,1024]
    float* out = (float*)d_out;                  // [4,2048,1024]

    char* ws = (char*)d_ws;
    _Float16* xh    = (_Float16*)(ws);                          // 16 MB
    _Float16* qkvh  = (_Float16*)(ws + (size_t)16777216);       // 48 MB
    _Float16* ctxh  = (_Float16*)(ws + (size_t)67108864);       // 16 MB
    _Float16* Wqkvh = (_Float16*)(ws + (size_t)83886080);       //  6 MB
    _Float16* Wouth = (_Float16*)(ws + (size_t)90177536);       //  2 MB

    // 1) f32 -> f16 conversions
    cvt_f32_to_f16<<<4096, 256, 0, stream>>>(x,    xh,    (size_t)ROWS * D_MODEL);
    cvt_f32_to_f16<<<2048, 256, 0, stream>>>(Wqkv, Wqkvh, (size_t)D_MODEL * QKV_COLS);
    cvt_f32_to_f16<<<1024, 256, 0, stream>>>(Wout, Wouth, (size_t)D_MODEL * D_MODEL);

    // 2) QKV projection: [8192,1024] @ [1024,3072] -> f16 [8192,3072]
    gemm_f16_wmma<_Float16><<<dim3(ROWS / 64, QKV_COLS / 128), 256, 0, stream>>>(
        xh, Wqkvh, qkvh, ROWS, QKV_COLS, D_MODEL);

    // 3) RoPE on q,k in place
    {
        const int total = ROWS * NUM_HEADS * (HEAD_DIM / 2);
        rope_kernel<<<(total + 255) / 256, 256, 0, stream>>>(qkvh);
    }

    // 4) Flash attention -> context f16 [8192,1024]
    flash_attn<<<dim3(BATCH * NUM_HEADS, SEQ / 64), 128, 0, stream>>>(qkvh, mask, ctxh);

    // 5) Output projection: [8192,1024] @ [1024,1024] -> f32 out
    gemm_f16_wmma<float><<<dim3(ROWS / 64, D_MODEL / 128), 256, 0, stream>>>(
        ctxh, Wouth, out, ROWS, D_MODEL, D_MODEL);
}
